// Attention_35777077575729
// MI455X (gfx1250) — matrix operand order repaired and checked by reading, not compiled, not run on hardware
//
#include <hip/hip_runtime.h>
#include <hip/hip_bf16.h>

// ---------------------------------------------------------------------------
// Flash attention for B=32, N=2048, D=128 (fp32 in/out, bf16 WMMA compute)
// gfx1250 / CDNA5: wave32, v_wmma_f32_16x16x32_bf16
//  - double-buffered LDS staging (global loads overlap WMMA)
//  - row-max reduction via DPP row_ror (VALU, no ds_bpermute)
//  - row-sum via one extra WMMA against a ones matrix
//  - native f32->bf16 converts (no manual bit twiddling)
// ---------------------------------------------------------------------------

typedef __bf16 bf16_t;
typedef __attribute__((ext_vector_type(16))) __bf16 v16bf;
typedef __attribute__((ext_vector_type(8)))  __bf16 bf8v;
typedef __attribute__((ext_vector_type(4)))  __bf16 bf4v;
typedef __attribute__((ext_vector_type(8)))  float  v8f;

#define N_SEQ   2048
#define D_DIM   128
#define QTILE   16            // query rows per wave
#define WAVES   8             // waves per block
#define BLOCK_Q (QTILE*WAVES) // 128 query rows per block
#define KTILE   32            // keys per inner iteration

// DPP move with compile-time control (row_ror within a 16-lane row)
template <int CTRL>
__device__ __forceinline__ float dpp_f(float x) {
    return __int_as_float(__builtin_amdgcn_update_dpp(
        0, __float_as_int(x), CTRL, 0xf, 0xf, true));
}
// max over the 16 lanes of this half-row; result valid in every lane
__device__ __forceinline__ float rowmax16(float x) {
    x = fmaxf(x, dpp_f<0x121>(x));   // row_ror:1
    x = fmaxf(x, dpp_f<0x122>(x));   // row_ror:2
    x = fmaxf(x, dpp_f<0x124>(x));   // row_ror:4
    x = fmaxf(x, dpp_f<0x128>(x));   // row_ror:8
    return x;
}

__global__ __launch_bounds__(256)
void fa_wmma_kernel(const float* __restrict__ q,
                    const float* __restrict__ k,
                    const float* __restrict__ v,
                    float* __restrict__ out)
{
    // Double-buffered: K row-major [key][d], V transposed [d][key]; per-wave P
    __shared__ __align__(32) bf16_t sK [2][KTILE * D_DIM];      // 2x8 KB
    __shared__ __align__(32) bf16_t sVt[2][D_DIM * KTILE];      // 2x8 KB
    __shared__ __align__(32) bf16_t sP [WAVES * QTILE * KTILE]; // 8 KB

    const int tid  = threadIdx.x;
    const int wave = tid >> 5;
    const int lane = tid & 31;
    const int h    = lane >> 4;   // half-wave id
    const int l16  = lane & 15;

    const int blk    = blockIdx.x;
    const int b      = blk / (N_SEQ / BLOCK_Q);
    const int q_base = (blk % (N_SEQ / BLOCK_Q)) * BLOCK_Q + wave * QTILE;

    const float scale = 0.08838834764831845f; // 1/sqrt(128)

    // ---- Q A-fragments: 16x32 bf16 per 32-wide d-chunk, pre-scaled --------
    v16bf qa[4];
    {
        const float* qrow = q + ((size_t)b * N_SEQ + q_base + l16) * D_DIM;
        #pragma unroll
        for (int c = 0; c < 4; ++c) {
            union { v16bf v; bf16_t e[16]; } u;
            #pragma unroll
            for (int i = 0; i < 8; ++i) {
                u.e[i]     = (bf16_t)(qrow[32*c + 16*h     + i] * scale);
                u.e[i + 8] = (bf16_t)(qrow[32*c + 16*h + 8 + i] * scale);
            }
            qa[c] = u.v;
        }
    }

    // ones B-fragment (32x16 bf16 of 1.0) for row-sum WMMA
    v16bf ones;
    {
        union { v16bf v; bf16_t e[16]; } u;
        #pragma unroll
        for (int i = 0; i < 16; ++i) u.e[i] = (bf16_t)1.0f;
        ones = u.v;
    }

    // ---- accumulators ------------------------------------------------------
    const v8f z8 = (v8f){0,0,0,0,0,0,0,0};
    v8f o[8];
    #pragma unroll
    for (int f = 0; f < 8; ++f) o[f] = z8;
    float mrow[8], lrow[8];          // row r of this half -> M = r + 8h
    #pragma unroll
    for (int r = 0; r < 8; ++r) { mrow[r] = -1e30f; lrow[r] = 0.0f; }

    const float* kbase = k + (size_t)b * N_SEQ * D_DIM;
    const float* vbase = v + (size_t)b * N_SEQ * D_DIM;
    bf16_t* pw = &sP[wave * QTILE * KTILE];

    // staging registers (software pipeline: loads in flight during WMMA)
    float4 kreg[4], vreg[4];

    auto issue_loads = [&](int kt_) {
        #pragma unroll
        for (int j = 0; j < 4; ++j) {
            const int e   = (tid + j * 256) * 4;
            const int key = e / D_DIM;
            const int d   = e % D_DIM;
            kreg[j] = *(const float4*)(kbase + (size_t)(kt_ + key) * D_DIM + d);
            vreg[j] = *(const float4*)(vbase + (size_t)(kt_ + key) * D_DIM + d);
        }
    };
    auto store_lds = [&](int buf_) {
        #pragma unroll
        for (int j = 0; j < 4; ++j) {
            const int e   = (tid + j * 256) * 4;
            const int key = e / D_DIM;
            const int d   = e % D_DIM;
            bf4v pk = { (bf16_t)kreg[j].x, (bf16_t)kreg[j].y,
                        (bf16_t)kreg[j].z, (bf16_t)kreg[j].w };
            *(bf4v*)&sK[buf_][key * D_DIM + d] = pk;
            sVt[buf_][(d + 0) * KTILE + key] = (bf16_t)vreg[j].x;
            sVt[buf_][(d + 1) * KTILE + key] = (bf16_t)vreg[j].y;
            sVt[buf_][(d + 2) * KTILE + key] = (bf16_t)vreg[j].z;
            sVt[buf_][(d + 3) * KTILE + key] = (bf16_t)vreg[j].w;
        }
    };

    // prologue: stage tile 0
    issue_loads(0);
    store_lds(0);
    __syncthreads();

    int buf = 0;
    for (int kt = 0; kt < N_SEQ; kt += KTILE) {
        const bool more = (kt + KTILE) < N_SEQ;
        if (more) issue_loads(kt + KTILE);            // HBM loads fly under WMMA
        if (kt + 2 * KTILE < N_SEQ) {                 // warm L2 for tile t+2
            __builtin_prefetch(kbase + (size_t)(kt + 2 * KTILE) * D_DIM + tid * 16, 0, 1);
            __builtin_prefetch(vbase + (size_t)(kt + 2 * KTILE) * D_DIM + tid * 16, 0, 1);
        }

        const bf16_t* cK  = &sK [buf][0];
        const bf16_t* cVt = &sVt[buf][0];

        // ---- S = Q . K^T : two 16x16 f32 accumulators (32 keys) ------------
        v16bf kb[8];
        #pragma unroll
        for (int c = 0; c < 4; ++c) {
            kb[2*c]     = *(const v16bf*)&cK[(     l16) * D_DIM + 32*c + 16*h];
            kb[2*c + 1] = *(const v16bf*)&cK[(16 + l16) * D_DIM + 32*c + 16*h];
        }
        v8f s0 = z8, s1 = z8;
        #pragma unroll
        for (int c = 0; c < 4; ++c) {
            s0 = __builtin_amdgcn_wmma_f32_16x16x32_bf16(false, qa[c], false, kb[2*c],
                                                         (short)0, s0, false, false);
            s1 = __builtin_amdgcn_wmma_f32_16x16x32_bf16(false, qa[c], false, kb[2*c+1],
                                                         (short)0, s1, false, false);
        }

        // ---- online softmax: DPP row-max, exp, rescale O -------------------
        float resc[8];
        #pragma unroll
        for (int r = 0; r < 8; ++r) {
            const float mx   = rowmax16(fmaxf(s0[r], s1[r]));
            const float mnew = fmaxf(mrow[r], mx);
            resc[r] = __expf(mrow[r] - mnew);
            mrow[r] = mnew;
            const float p0 = __expf(s0[r] - mnew);
            const float p1 = __expf(s1[r] - mnew);
            const int M = r + 8*h;                    // lanes 0-15 -> key = l16
            pw[M * KTILE +      l16] = (bf16_t)p0;
            pw[M * KTILE + 16 + l16] = (bf16_t)p1;
            #pragma unroll
            for (int f = 0; f < 8; ++f) o[f][r] *= resc[r];
        }

        // ---- re-layout P as A-fragment (wave-private LDS round trip) -------
        union { v16bf v; bf8v g[2]; } up;
        up.g[0] = *(const bf8v*)&pw[l16 * KTILE + 16*h];
        up.g[1] = *(const bf8v*)&pw[l16 * KTILE + 16*h + 8];
        const v16bf pa = up.v;

        // ---- row sums via WMMA against ones (lands in matching C layout) ---
        const v8f rs = __builtin_amdgcn_wmma_f32_16x16x32_bf16(false, pa, false, ones,
                                                               (short)0, z8, false, false);
        #pragma unroll
        for (int r = 0; r < 8; ++r) lrow[r] = lrow[r] * resc[r] + rs[r];

        // ---- O += P . V (8 output d-chunks of 16) --------------------------
        v16bf vb[8];
        #pragma unroll
        for (int f = 0; f < 8; ++f)
            vb[f] = *(const v16bf*)&cVt[(16*f + l16) * KTILE + 16*h];
        #pragma unroll
        for (int f = 0; f < 8; ++f)
            o[f] = __builtin_amdgcn_wmma_f32_16x16x32_bf16(false, pa, false, vb[f],
                                                           (short)0, o[f], false, false);

        if (more) store_lds(buf ^ 1);                 // commit next tile to LDS
        __syncthreads();
        buf ^= 1;
    }

    // ---- epilogue: O /= l, write fp32 --------------------------------------
    float inv[8];
    #pragma unroll
    for (int r = 0; r < 8; ++r) inv[r] = 1.0f / lrow[r];

    float* obase = out + ((size_t)b * N_SEQ + q_base) * D_DIM;
    #pragma unroll
    for (int f = 0; f < 8; ++f) {
        const int d = 16*f + l16;
        #pragma unroll
        for (int r = 0; r < 8; ++r) {
            const int M = r + 8*h;
            obase[(size_t)M * D_DIM + d] = o[f][r] * inv[r];
        }
    }
}

extern "C" void kernel_launch(void* const* d_in, const int* in_sizes, int n_in,
                              void* d_out, int out_size, void* d_ws, size_t ws_size,
                              hipStream_t stream) {
    (void)in_sizes; (void)n_in; (void)out_size; (void)d_ws; (void)ws_size;
    const float* q = (const float*)d_in[0];
    const float* k = (const float*)d_in[1];
    const float* v = (const float*)d_in[2];
    float* out = (float*)d_out;

    dim3 grid(32 * (N_SEQ / BLOCK_Q));  // 32 batches * 16 q-chunks = 512 blocks
    dim3 block(256);                    // 8 waves (wave32)
    fa_wmma_kernel<<<grid, block, 0, stream>>>(q, k, v, out);
}
